// DiffPOI_31499290149126
// MI455X (gfx1250) — matrix operand order repaired and checked by reading, not compiled, not run on hardware
//
#include <hip/hip_runtime.h>
#include <hip/hip_bf16.h>

#define HID 64
#define LEAKY 0.01f

typedef __attribute__((ext_vector_type(16))) __bf16 v16bf;
typedef __attribute__((ext_vector_type(8)))  __bf16 v8bf;
typedef __attribute__((ext_vector_type(8)))  float  v8f;

// ---------- helpers ----------
__device__ __forceinline__ __bf16 f2bf(float f) {
    union { float f; unsigned u; } v; v.f = f;
    unsigned r = v.u + 0x7FFFu + ((v.u >> 16) & 1u);  // RNE
    unsigned short h = (unsigned short)(r >> 16);
    __bf16 b; __builtin_memcpy(&b, &h, 2);
    return b;
}

__device__ __forceinline__ void atomicMaxF(float* addr, float val) {
    if (val >= 0.0f) atomicMax((int*)addr, __float_as_int(val));
    else             atomicMin((unsigned int*)addr, (unsigned int)__float_as_int(val));
}

// =====================================================================
// Sequence-graph attention (seq_conv)
// =====================================================================

__global__ void seq_init_kernel(float* out_seq, float* m, float* denom, int N) {
    int i = blockIdx.x * blockDim.x + threadIdx.x;
    if (i < N * HID) out_seq[i] = 0.0f;
    if (i < N) { m[i] = -__builtin_inff(); denom[i] = 0.0f; }
}

// x[n] = poi_emb[sess_idx[n]]   (float4 per thread, coalesced)
__global__ void seq_gather_kernel(const float* __restrict__ poi,
                                  const int* __restrict__ sidx,
                                  float* __restrict__ x, int N) {
    int i = blockIdx.x * blockDim.x + threadIdx.x;   // N*16 threads
    if (i >= N * (HID / 4)) return;
    int n = i >> 4, c = i & 15;
    ((float4*)(x + (size_t)n * HID))[c] =
        ((const float4*)(poi + (size_t)sidx[n] * HID))[c];
}

// one wave per edge: both direction logits + atomic segment-max
__global__ void seq_logits_kernel(const float* __restrict__ x,
                                  const float* __restrict__ demb,
                                  const float* __restrict__ temb,
                                  const float* __restrict__ a_src,
                                  const float* __restrict__ a_dst,
                                  const int* __restrict__ sei,
                                  const int* __restrict__ etime,
                                  const int* __restrict__ edist,
                                  float* __restrict__ logits,
                                  float* __restrict__ m, int E) {
    int e = blockIdx.x * (blockDim.x >> 5) + (threadIdx.x >> 5);
    if (e >= E) return;
    int lane = threadIdx.x & 31;
    int src = sei[e], dst = sei[E + e];
    float2 vs = ((const float2*)(x + (size_t)src * HID))[lane];
    float2 vd = ((const float2*)(x + (size_t)dst * HID))[lane];
    float2 el = ((const float2*)(demb + (size_t)edist[e] * HID))[lane];
    float2 et = ((const float2*)(temb + (size_t)etime[e] * HID))[lane];
    float sx = vs.x * vd.x + el.x + et.x;
    float sy = vs.y * vd.y + el.y + et.y;
    float2 as_ = ((const float2*)a_src)[lane];
    float2 ad_ = ((const float2*)a_dst)[lane];
    float ls = sx * as_.x + sy * as_.y;   // target = dst
    float ld = sx * ad_.x + sy * ad_.y;   // target = src
    #pragma unroll
    for (int off = 16; off; off >>= 1) {
        ls += __shfl_xor(ls, off);
        ld += __shfl_xor(ld, off);
    }
    if (lane == 0) {
        logits[e]     = ls;
        logits[E + e] = ld;
        atomicMaxF(&m[dst], ls);
        atomicMaxF(&m[src], ld);
    }
}

// ex = exp(logit - m[target]); denom[target] += ex
__global__ void seq_expsum_kernel(const int* __restrict__ sei,
                                  const float* __restrict__ logits,
                                  const float* __restrict__ m,
                                  float* __restrict__ ex,
                                  float* __restrict__ denom, int E) {
    int e2 = blockIdx.x * blockDim.x + threadIdx.x;
    if (e2 >= 2 * E) return;
    int i = sei[(e2 + E) % (2 * E)];         // target
    float v = __expf(logits[e2] - m[i]);
    ex[e2] = v;
    atomicAdd(&denom[i], v);
}

// out[i] += attn * x[j]   (wave per message, lane per float2)
__global__ void seq_scatter_kernel(const int* __restrict__ sei,
                                   const float* __restrict__ x,
                                   const float* __restrict__ ex,
                                   const float* __restrict__ denom,
                                   float* __restrict__ out, int E) {
    int e2 = blockIdx.x * (blockDim.x >> 5) + (threadIdx.x >> 5);
    if (e2 >= 2 * E) return;
    int lane = threadIdx.x & 31;
    int j = sei[e2];
    int i = sei[(e2 + E) % (2 * E)];
    float a = ex[e2] / (denom[i] + 1e-16f);
    float2 xv = ((const float2*)(x + (size_t)j * HID))[lane];
    float* dst = out + (size_t)i * HID + lane * 2;
    atomicAdd(dst,     xv.x * a);
    atomicAdd(dst + 1, xv.y * a);
}

// =====================================================================
// Geo GCN encoder
// =====================================================================

// layer_bf = bf16(poi_emb); out_geo = poi_emb; dinv = 1 (self-loop degree)
__global__ void geo_init_kernel(const float* __restrict__ poi,
                                __bf16* __restrict__ layer_bf,
                                float* __restrict__ out_geo,
                                float* __restrict__ dinv, int Npoi) {
    int i = blockIdx.x * blockDim.x + threadIdx.x;
    if (i < Npoi * HID) {
        float v = poi[i];
        layer_bf[i] = f2bf(v);
        out_geo[i] = v;
    }
    if (i < Npoi) dinv[i] = 1.0f;
}

__global__ void wconv_kernel(const float* __restrict__ W, __bf16* __restrict__ wbf, int n) {
    int i = blockIdx.x * blockDim.x + threadIdx.x;
    if (i < n) wbf[i] = f2bf(W[i]);
}

__global__ void geo_deg_kernel(const int* __restrict__ gei, float* __restrict__ deg, int E) {
    int e = blockIdx.x * blockDim.x + threadIdx.x;
    if (e < E) atomicAdd(&deg[gei[E + e]], 1.0f);
}

__global__ void geo_dinv_kernel(float* __restrict__ dinv, int Npoi) {
    int i = blockIdx.x * blockDim.x + threadIdx.x;
    if (i < Npoi) dinv[i] = __frsqrt_rn(dinv[i]);   // deg >= 1 always
}

__global__ void zero_kernel(float* __restrict__ p, int n) {
    int i = blockIdx.x * blockDim.x + threadIdx.x;
    if (i < n) p[i] = 0.0f;
}

// h = layer @ W^T + b  via v_wmma_f32_16x16x32_bf16.
// Block = 128 threads (4 waves). Wave w -> N-tile w (16 cols). Block -> 16 rows.
// A and B are both bf16 in memory -> pure vector loads, zero conversion VALU.
__global__ void geo_gemm_wmma(const __bf16* __restrict__ layer_bf,  // [M,64] bf16
                              const __bf16* __restrict__ wbf,       // W[l]: [n][k] bf16
                              const float* __restrict__ bias,
                              float* __restrict__ h, int M) {
    int wave = threadIdx.x >> 5;            // n-tile 0..3
    int lane = threadIdx.x & 31;
    int hi   = lane >> 4;                   // lane half
    int m_tile = blockIdx.x << 4;
    int n0 = wave << 4;
    int row = m_tile + (lane & 15);
    int n   = n0 + (lane & 15);

    v8f acc = {};
    #pragma unroll
    for (int kb = 0; kb < HID; kb += 32) {
        // A 16x32 bf16: lane row = lane%16; needs K = kb+hi*16 + {0..15}
        const __bf16* Arow = layer_bf + (size_t)row * HID + kb + hi * 16;
        v8bf a_lo = *(const v8bf*)(Arow);
        v8bf a_hh = *(const v8bf*)(Arow + 8);
        v16bf A = __builtin_shufflevector(a_lo, a_hh,
                      0, 1, 2, 3, 4, 5, 6, 7, 8, 9, 10, 11, 12, 13, 14, 15);
        // B 32x16 bf16: B[K][N] = W[n][k]; lane needs 16 contiguous k at kb+hi*16
        v16bf B = *(const v16bf*)(wbf + (size_t)n * HID + kb + hi * 16);
        acc = __builtin_amdgcn_wmma_f32_16x16x32_bf16(
                  false, A, false, B, (short)0, acc, false, false);
    }
    float bn = bias[n];
    #pragma unroll
    for (int r = 0; r < 8; ++r) {
        int mm = m_tile + hi * 8 + r;       // C/D layout: VGPR r -> row hi*8+r
        h[(size_t)mm * HID + n] = acc[r] + bn;
    }
}

// agg[col] += dinv[row]*dinv[col]*exp(-d^2) * h[row]  (wave per edge incl. self loops)
__global__ void geo_scatter_kernel(const int* __restrict__ gei,
                                   const float* __restrict__ gdist,
                                   const float* __restrict__ dinv,
                                   const float* __restrict__ h,
                                   float* __restrict__ agg, int E, int Npoi) {
    int e = blockIdx.x * (blockDim.x >> 5) + (threadIdx.x >> 5);
    if (e >= E + Npoi) return;
    int lane = threadIdx.x & 31;
    int r, c; float w;
    if (e < E) { r = gei[e]; c = gei[E + e]; float d = gdist[e]; w = __expf(-d * d); }
    else       { r = c = e - E; w = 1.0f; }
    float coef = dinv[r] * dinv[c] * w;
    float2 hv = ((const float2*)(h + (size_t)r * HID))[lane];
    float* dst = agg + (size_t)c * HID + lane * 2;
    atomicAdd(dst,     hv.x * coef);
    atomicAdd(dst + 1, hv.y * coef);
}

// v = leaky(agg); layer_bf = bf16(v); out_geo += v; final layer divides by 3
__global__ void relu_acc_kernel(const float* __restrict__ agg,
                                __bf16* __restrict__ layer_bf,
                                float* __restrict__ out_geo, int n, int finalFlag) {
    int i = blockIdx.x * blockDim.x + threadIdx.x;
    if (i >= n) return;
    float v = agg[i];
    v = (v >= 0.0f) ? v : LEAKY * v;
    layer_bf[i] = f2bf(v);
    float acc = out_geo[i] + v;
    out_geo[i] = finalFlag ? acc * (1.0f / 3.0f) : acc;
}

// =====================================================================
extern "C" void kernel_launch(void* const* d_in, const int* in_sizes, int n_in,
                              void* d_out, int out_size, void* d_ws, size_t ws_size,
                              hipStream_t stream) {
    const float* poi_emb  = (const float*)d_in[0];
    const float* demb     = (const float*)d_in[1];
    const float* temb     = (const float*)d_in[2];
    const float* a_src    = (const float*)d_in[3];
    const float* a_dst    = (const float*)d_in[4];
    const float* W_geo    = (const float*)d_in[5];
    const float* b_geo    = (const float*)d_in[6];
    const float* geo_dist = (const float*)d_in[7];
    const int*   gei      = (const int*)d_in[8];
    const int*   sidx     = (const int*)d_in[9];
    const int*   sei      = (const int*)d_in[10];
    const int*   etime    = (const int*)d_in[11];
    const int*   edist    = (const int*)d_in[12];

    const int n_poi = in_sizes[0] / HID;
    const int E_geo = in_sizes[7];
    const int N     = in_sizes[9];
    const int E     = in_sizes[10] / 2;

    float* out_seq = (float*)d_out;
    float* out_geo = out_seq + (size_t)N * HID;

    // workspace: seq phase and geo phase alias the same arena (sequential)
    float* x      = (float*)d_ws;                 // N*HID
    float* logits = x + (size_t)N * HID;          // 2E
    float* ex     = logits + (size_t)2 * E;       // 2E
    float* mbuf   = ex + (size_t)2 * E;           // N
    float* denom  = mbuf + N;                     // N

    float*  dinv     = (float*)d_ws;              // n_poi
    float*  h        = dinv + n_poi;              // n_poi*HID f32
    float*  agg      = h + (size_t)n_poi * HID;   // n_poi*HID f32
    __bf16* layer_bf = (__bf16*)(agg + (size_t)n_poi * HID); // n_poi*HID bf16
    __bf16* wbf      = layer_bf + (size_t)n_poi * HID;       // 2*HID*HID bf16

    const int B = 256;
    // ---------------- seq phase ----------------
    seq_init_kernel<<<(N * HID + B - 1) / B, B, 0, stream>>>(out_seq, mbuf, denom, N);
    seq_gather_kernel<<<(N * (HID / 4) + B - 1) / B, B, 0, stream>>>(poi_emb, sidx, x, N);
    seq_logits_kernel<<<(E + 7) / 8, B, 0, stream>>>(x, demb, temb, a_src, a_dst,
                                                     sei, etime, edist, logits, mbuf, E);
    seq_expsum_kernel<<<(2 * E + B - 1) / B, B, 0, stream>>>(sei, logits, mbuf, ex, denom, E);
    seq_scatter_kernel<<<(2 * E + 7) / 8, B, 0, stream>>>(sei, x, ex, denom, out_seq, E);

    // ---------------- geo phase ----------------
    geo_init_kernel<<<(n_poi * HID + B - 1) / B, B, 0, stream>>>(poi_emb, layer_bf, out_geo, dinv, n_poi);
    wconv_kernel<<<(2 * HID * HID + B - 1) / B, B, 0, stream>>>(W_geo, wbf, 2 * HID * HID);
    geo_deg_kernel<<<(E_geo + B - 1) / B, B, 0, stream>>>(gei, dinv, E_geo);
    geo_dinv_kernel<<<(n_poi + B - 1) / B, B, 0, stream>>>(dinv, n_poi);

    for (int l = 0; l < 2; ++l) {
        zero_kernel<<<(n_poi * HID + B - 1) / B, B, 0, stream>>>(agg, n_poi * HID);
        geo_gemm_wmma<<<n_poi / 16, 128, 0, stream>>>(layer_bf, wbf + (size_t)l * HID * HID,
                                                      b_geo + (size_t)l * HID, h, n_poi);
        geo_scatter_kernel<<<(E_geo + n_poi + 7) / 8, B, 0, stream>>>(gei, geo_dist, dinv,
                                                                      h, agg, E_geo, n_poi);
        relu_acc_kernel<<<(n_poi * HID + B - 1) / B, B, 0, stream>>>(agg, layer_bf, out_geo,
                                                                     n_poi * HID, l == 1);
    }
}
